// MEResUNet_76381698392648
// MI455X (gfx1250) — matrix operand order repaired and checked by reading, not compiled, not run on hardware
//
#include <hip/hip_runtime.h>

// ---------------------------------------------------------------------------
// MinkowskiEngine-style sparse ResUNet on gfx1250 (MI455X).
// Sparse conv = per-offset gather -> f16 WMMA GEMM (f32 accum) -> atomic scatter.
// NT=2 column-tile blocking: each wave reuses its gathered A fragment for two
// 16-wide cout tiles (all couts are multiples of 32), halving gather traffic.
// ---------------------------------------------------------------------------

typedef __attribute__((ext_vector_type(16))) _Float16 v16h;
typedef __attribute__((ext_vector_type(8)))  _Float16 h8v;
typedef __attribute__((ext_vector_type(8)))  float    v8f;

__device__ __forceinline__ v16h pack16(h8v lo, h8v hi) {
  return __builtin_shufflevector(lo, hi, 0, 1, 2, 3, 4, 5, 6, 7,
                                 8, 9, 10, 11, 12, 13, 14, 15);
}

// ----------------------------- utility kernels -----------------------------

__global__ void zero_f32_k(float* __restrict__ p, long long n) {
  long long i = (long long)blockIdx.x * blockDim.x + threadIdx.x;
  if (i < n) p[i] = 0.f;
}

// w: [27][cin][cout] fp32  ->  o: [27][cout][cinpad] f16 (zero-padded K)
__global__ void cvt_w_sparse_k(const float* __restrict__ w, int cin, int cout,
                               int cinpad, _Float16* __restrict__ o) {
  long long tot = 27LL * cout * cinpad;
  long long i = (long long)blockIdx.x * blockDim.x + threadIdx.x;
  if (i >= tot) return;
  int r = (int)(i % cinpad);
  long long t = i / cinpad;
  int c = (int)(t % cout);
  int k = (int)(t / cout);
  o[i] = (r < cin) ? (_Float16)w[((long long)k * cin + r) * cout + c] : (_Float16)0.f;
}

// w: [cin][cout] fp32 -> o: [cout][cinpad] f16
__global__ void cvt_w_dense_k(const float* __restrict__ w, int cin, int cout,
                              int cinpad, _Float16* __restrict__ o) {
  long long tot = (long long)cout * cinpad;
  long long i = (long long)blockIdx.x * blockDim.x + threadIdx.x;
  if (i >= tot) return;
  int r = (int)(i % cinpad);
  int c = (int)(i / cinpad);
  o[i] = (r < cin) ? (_Float16)w[(long long)r * cout + c] : (_Float16)0.f;
}

// x (n x 1) -> act16 (n+1 rows x 32 cols, pre-zeroed), col 0
__global__ void init_x16_k(const float* __restrict__ x, int n, _Float16* __restrict__ act) {
  int i = blockIdx.x * blockDim.x + threadIdx.x;
  if (i < n) act[(long long)i * 32] = (_Float16)x[i];
}

// fp32 (n x c) -> f16 buffer with leading dim ld at column offset (for concat)
__global__ void f32_to_f16_k(const float* __restrict__ x, int n, int c,
                             _Float16* __restrict__ y, int ld, int coloff) {
  long long i = (long long)blockIdx.x * blockDim.x + threadIdx.x;
  if (i >= (long long)n * c) return;
  int col = (int)(i % c);
  long long row = i / c;
  y[row * ld + coloff + col] = (_Float16)x[i];
}

// per-channel sum / sumsq via atomics (stats pre-zeroed, length 2*c)
__global__ void bn_stats_k(const float* __restrict__ x, int n, int c,
                           float* __restrict__ st) {
  int col = (blockIdx.y << 5) + (threadIdx.x & 31);
  int row0 = blockIdx.x * (blockDim.x >> 5) + (threadIdx.x >> 5);
  int stride = gridDim.x * (blockDim.x >> 5);
  if (col >= c) return;
  float s = 0.f, q = 0.f;
  for (long long r = row0; r < n; r += stride) {
    float v = x[r * c + col];
    s += v; q += v * v;
  }
  atomicAdd(&st[col], s);
  atomicAdd(&st[c + col], q);
}

// y = g*(x-mu)*rsqrt(var+eps)+b [+res] [relu]; writes fp32 and/or f16 copy
__global__ void bn_apply_k(const float* __restrict__ xin, const float* __restrict__ st,
                           const float* __restrict__ g, const float* __restrict__ b,
                           const float* __restrict__ res, float invn, int n, int c,
                           int relu, float* __restrict__ yf,
                           _Float16* __restrict__ y16, int ld16, int coloff) {
  long long i = (long long)blockIdx.x * blockDim.x + threadIdx.x;
  if (i >= (long long)n * c) return;
  int col = (int)(i % c);
  long long row = i / c;
  float mu = st[col] * invn;
  float var = st[c + col] * invn - mu * mu;
  if (var < 0.f) var = 0.f;
  float y = g[col] * (xin[i] - mu) * rsqrtf(var + 1e-5f) + b[col];
  if (res) y += res[i];
  if (relu) y = fmaxf(y, 0.f);
  if (yf) yf[i] = y;
  if (y16) y16[row * ld16 + coloff + col] = (_Float16)y;
}

// final: out = x / max(||x||_2, 1e-12), c == 32, one wave per row
__global__ void l2norm_k(const float* __restrict__ x, int n, float* __restrict__ out) {
  int row = blockIdx.x * 8 + (threadIdx.x >> 5);
  int lane = threadIdx.x & 31;
  if (row >= n) return;
  float v = x[(long long)row * 32 + lane];
  float s = v * v;
#pragma unroll
  for (int o = 16; o; o >>= 1) s += __shfl_xor(s, o, 32);
  float nrm = sqrtf(s);
  if (nrm < 1e-12f) nrm = 1e-12f;
  out[(long long)row * 32 + lane] = v / nrm;
}

// ----------------------------- WMMA kernels --------------------------------
// Fragment layouts used here (wave32; contiguous K, not the ISA table order):
//  A (16x32 f16): row = lane&15; lane<16 K 0..15, lane>=16 K 16..31.
//  B (32x16 f16): col = lane&15; lane<16 K 0..15, lane>=16 K 16..31.
//  C/D (16x16 f32): col = lane&15; vgpr r -> row r + 8*(lane>>4).

// sparse conv: out[ko[k][p]] += (fin[ki[k][p]] @ W[k]) with atomic scatter.
// fin: (nin+1) x cinpad f16, row nin is zeros. wt: [27][cout][cinpad] f16.
// Each wave: 16 pairs x 32 cout columns (two 16x16 WMMA tiles sharing A).
__global__ __launch_bounds__(128) void sconv16_k(
    const _Float16* __restrict__ fin, int cinpad, const _Float16* __restrict__ wt,
    const int* __restrict__ ki, const int* __restrict__ ko, int P, int nin, int nout,
    int cout, float* __restrict__ out) {
  extern __shared__ _Float16 ldsS[];  // 32 x cinpad (N-major, K contiguous)
  int k = blockIdx.z;
  int nblk = blockIdx.y;              // 32-wide cout tile
  int tid = threadIdx.x, lane = tid & 31, wave = tid >> 5;

  // stage W tile for this (k, nblk): contiguous 32*cinpad halves
  {
    const h8v* src = (const h8v*)(wt + ((long long)k * cout + (long long)nblk * 32) * cinpad);
    h8v* dst = (h8v*)ldsS;
    int tot8 = (cinpad * 32) >> 3;
    for (int i = tid; i < tot8; i += blockDim.x) dst[i] = src[i];
  }
  __syncthreads();

  int row0 = (blockIdx.x * 4 + wave) * 16;
  if (row0 >= P) return;  // uniform per wave

  const int* kiK = ki + (long long)k * P;
  const int* koK = ko + (long long)k * P;

  int m = lane & 15;
  int pr = row0 + m;
  int gsrc = (pr < P) ? kiK[pr] : nin;
  if ((unsigned)gsrc > (unsigned)nin) gsrc = nin;  // dummy/pad -> zero row
  const _Float16* arow = fin + (long long)gsrc * cinpad;
  const int kbA = (lane & 16) ? 16 : 0;
  const int kbB = (lane & 16) ? 16 : 0;
  const _Float16* brow0 = ldsS + (long long)m * cinpad + kbB;          // cols 0..15
  const _Float16* brow1 = ldsS + (long long)(m + 16) * cinpad + kbB;   // cols 16..31

  v8f acc0 = {};
  v8f acc1 = {};
  for (int kc = 0; kc < cinpad; kc += 32) {
    v16h a = pack16(*(const h8v*)(arow + kc + kbA),
                    *(const h8v*)(arow + kc + kbA + 8));
    v16h b0 = pack16(*(const h8v*)(brow0 + kc), *(const h8v*)(brow0 + kc + 8));
    v16h b1 = pack16(*(const h8v*)(brow1 + kc), *(const h8v*)(brow1 + kc + 8));
    acc0 = __builtin_amdgcn_wmma_f32_16x16x32_f16(false, a, false, b0,
                                                  (short)0, acc0, false, false);
    acc1 = __builtin_amdgcn_wmma_f32_16x16x32_f16(false, a, false, b1,
                                                  (short)0, acc1, false, false);
  }

  int colg = nblk * 32 + m;
#pragma unroll
  for (int r = 0; r < 8; ++r) {
    int mr = r + ((lane >> 4) << 3);
    int prr = row0 + mr;
    if (prr < P) {
      int dst = koK[prr];
      if ((unsigned)dst < (unsigned)nout) {
        float* orow = out + (long long)dst * cout + colg;
        atomicAdd(orow, acc0[r]);
        atomicAdd(orow + 16, acc1[r]);
      }
    }
  }
}

// dense 1x1 GEMM: out[n x cout] = A[n x cinpad] @ W, optional bias/relu,
// optional f16 copy of the result. A has a zero row at index n.
// NT=2 tiles of 16 cols per wave, always; cout must be a multiple of 32 (cout 32/64 here -> 2
// when possible; cout==32 uses gridDim.y==1 with the full 32 cols).
__global__ __launch_bounds__(128) void dgemm16_k(
    const _Float16* __restrict__ A, int cinpad, const _Float16* __restrict__ wt,
    int cout, int n, const float* __restrict__ bias, int relu,
    float* __restrict__ outf, _Float16* __restrict__ out16, int ld16) {
  extern __shared__ _Float16 ldsD[];
  int nblk = blockIdx.y;              // 32-wide cout tile
  int tid = threadIdx.x, lane = tid & 31, wave = tid >> 5;
  {
    const h8v* src = (const h8v*)(wt + (long long)nblk * 32 * cinpad);
    h8v* dst = (h8v*)ldsD;
    int tot8 = (cinpad * 32) >> 3;
    for (int i = tid; i < tot8; i += blockDim.x) dst[i] = src[i];
  }
  __syncthreads();

  int row0 = (blockIdx.x * 4 + wave) * 16;
  if (row0 >= n) return;

  int m = lane & 15;
  int rin = row0 + m;
  if (rin > n) rin = n;  // zero row
  const _Float16* arow = A + (long long)rin * cinpad;
  const int kbA = (lane & 16) ? 16 : 0;
  const int kbB = (lane & 16) ? 16 : 0;
  const _Float16* brow0 = ldsD + (long long)m * cinpad + kbB;
  const _Float16* brow1 = ldsD + (long long)(m + 16) * cinpad + kbB;

  v8f acc0 = {};
  v8f acc1 = {};
  for (int kc = 0; kc < cinpad; kc += 32) {
    v16h a = pack16(*(const h8v*)(arow + kc + kbA),
                    *(const h8v*)(arow + kc + kbA + 8));
    v16h b0 = pack16(*(const h8v*)(brow0 + kc), *(const h8v*)(brow0 + kc + 8));
    v16h b1 = pack16(*(const h8v*)(brow1 + kc), *(const h8v*)(brow1 + kc + 8));
    acc0 = __builtin_amdgcn_wmma_f32_16x16x32_f16(false, a, false, b0,
                                                  (short)0, acc0, false, false);
    acc1 = __builtin_amdgcn_wmma_f32_16x16x32_f16(false, a, false, b1,
                                                  (short)0, acc1, false, false);
  }

  int col = nblk * 32 + m;
  float bv0 = bias ? bias[col] : 0.f;
  float bv1 = bias ? bias[col + 16] : 0.f;
#pragma unroll
  for (int r = 0; r < 8; ++r) {
    int row = row0 + r + ((lane >> 4) << 3);
    if (row < n) {
      float v0 = acc0[r] + bv0;
      float v1 = acc1[r] + bv1;
      if (relu) { v0 = fmaxf(v0, 0.f); v1 = fmaxf(v1, 0.f); }
      if (outf) {
        outf[(long long)row * cout + col] = v0;
        outf[(long long)row * cout + col + 16] = v1;
      }
      if (out16) {
        out16[(long long)row * ld16 + col] = (_Float16)v0;
        out16[(long long)row * ld16 + col + 16] = (_Float16)v1;
      }
    }
  }
}

// ----------------------------- host helpers --------------------------------

static inline void zero_f(hipStream_t s, void* p, size_t elems_f32) {
  if (!elems_f32) return;
  long long n = (long long)elems_f32;
  unsigned g = (unsigned)((n + 255) / 256);
  zero_f32_k<<<dim3(g), dim3(256), 0, s>>>((float*)p, n);
}
static inline void zero_h(hipStream_t s, void* p, size_t halves) {
  zero_f(s, p, halves / 2);  // half counts are always even here
}

static void run_sconv(hipStream_t s, const _Float16* act, int cinpad, const float* W,
                      int cin, int cout, _Float16* w16, const int* ki, const int* ko,
                      int P, int nin, int nout, float* tmp) {
  long long tw = 27LL * cout * cinpad;
  cvt_w_sparse_k<<<dim3((unsigned)((tw + 255) / 256)), dim3(256), 0, s>>>(
      W, cin, cout, cinpad, w16);
  zero_f(s, tmp, (size_t)nout * cout);
  dim3 grid((unsigned)((P + 63) / 64), (unsigned)(cout / 32), 27);
  sconv16_k<<<grid, dim3(128), (size_t)32 * cinpad * sizeof(_Float16), s>>>(
      act, cinpad, w16, ki, ko, P, nin, nout, cout, tmp);
}

static void run_bn(hipStream_t s, const float* xin, const float* g, const float* b,
                   const float* res, int n, int c, int relu, float* yf32,
                   _Float16* y16, int ld16, int coloff, float* stats) {
  zero_f(s, stats, (size_t)2 * c);
  bn_stats_k<<<dim3(128, (unsigned)((c + 31) / 32)), dim3(256), 0, s>>>(xin, n, c, stats);
  long long tot = (long long)n * c;
  bn_apply_k<<<dim3((unsigned)((tot + 255) / 256)), dim3(256), 0, s>>>(
      xin, stats, g, b, res, 1.0f / (float)n, n, c, relu, yf32, y16, ld16, coloff);
}

// ----------------------------- orchestration -------------------------------
// Input leaves follow setup_inputs() dict insertion order:
//   0: x
//   1..66: params leaves in insertion order (each conv block: W, g, b):
//     conv1(1,2,3) b1a(4..6) b1b(7..9) conv2(10..12) b2a(13..15) b2b(16..18)
//     conv3(19..21) b3a(22..24) b3b(25..27) conv4(28..30) b4a(31..33) b4b(34..36)
//     conv4_tr(37..39) b4ta(40..42) b4tb(43..45) conv3_tr(46..48) b3ta(49..51)
//     b3tb(52..54) conv2_tr(55..57) b2ta(58..60) b2tb(61..63)
//     conv1_tr.W(64) final.W(65) final.b(66)
//   67..80: km_s1_in/out, km_s2_in/out, km_s4_in/out, km_s8_in/out,
//           kd1_in/out, kd2_in/out, kd3_in/out   (each int32, 27*P)
//   81..83: z1, z2, z3 (row counts n1, n2, n3)

extern "C" void kernel_launch(void* const* d_in, const int* in_sizes, int n_in,
                              void* d_out, int out_size, void* d_ws, size_t ws_size,
                              hipStream_t stream) {
  (void)n_in; (void)out_size; (void)ws_size;
  auto F = [&](int i) { return (const float*)d_in[i]; };
  auto I = [&](int i) { return (const int*)d_in[i]; };

  const float* x = F(0);
  const int n0 = in_sizes[0];
  const int n1 = in_sizes[81], n2 = in_sizes[82], n3 = in_sizes[83];
  const int Ps1 = in_sizes[67] / 27, Ps2 = in_sizes[69] / 27;
  const int Ps4 = in_sizes[71] / 27, Ps8 = in_sizes[73] / 27;
  const int Pd1 = in_sizes[75] / 27, Pd2 = in_sizes[77] / 27, Pd3 = in_sizes[79] / 27;

  auto mx = [](size_t a, size_t b) { return a > b ? a : b; };

  // workspace layout (256B aligned)
  char* ws = (char*)d_ws;
  size_t off = 0;
  auto alloc = [&](size_t bytes) -> void* {
    void* p = ws + off;
    off = (off + bytes + 255) & ~(size_t)255;
    return p;
  };
  float* S1 = (float*)alloc((size_t)n0 * 32 * 4);
  float* S2 = (float*)alloc((size_t)n1 * 64 * 4);
  float* S4 = (float*)alloc((size_t)n2 * 128 * 4);
  float* S8 = (float*)alloc((size_t)n3 * 256 * 4);
  float* T4 = (float*)alloc((size_t)n2 * 128 * 4);
  float* T2 = (float*)alloc((size_t)n1 * 64 * 4);
  float* T1 = (float*)alloc((size_t)n0 * 64 * 4);
  size_t maxTmp = mx((size_t)n0 * 64,
                  mx((size_t)n1 * 64, mx((size_t)n2 * 128, (size_t)n3 * 256)));
  float* TMP = (float*)alloc(maxTmp * 4);
  float* STATS = (float*)alloc(512 * 4);
  size_t maxAct = 0;
  maxAct = mx(maxAct, (size_t)(n0 + 1) * 96);
  maxAct = mx(maxAct, (size_t)(n1 + 1) * 128);
  maxAct = mx(maxAct, (size_t)(n2 + 1) * 256);
  maxAct = mx(maxAct, (size_t)(n3 + 1) * 256);
  _Float16* A0 = (_Float16*)alloc(maxAct * 2);
  _Float16* A1 = (_Float16*)alloc(maxAct * 2);
  _Float16* W16 = (_Float16*)alloc((size_t)27 * 256 * 256 * 2);

  _Float16* cur = A0;
  _Float16* nxt = A1;
  auto swapact = [&]() { _Float16* t = cur; cur = nxt; nxt = t; };

  // conv + bn step: tmp = sconv(cur); bn(tmp) -> yf32 (opt) and f16 into nxt.
  auto cbr = [&](int cinpad, int wi, int cin, int cout, const int* ki, const int* ko,
                 int P, int nin, int nout, const float* res, int relu, float* yf32,
                 int nextLd) {
    run_sconv(stream, cur, cinpad, F(wi), cin, cout, W16, ki, ko, P, nin, nout, TMP);
    zero_h(stream, nxt, (size_t)(nout + 1) * nextLd);
    run_bn(stream, TMP, F(wi + 1), F(wi + 2), res, nout, cout, relu, yf32,
           nxt, nextLd, 0, STATS);
    swapact();
  };
  auto concat = [&](const float* src, int n, int c, int ld, int coloff) {
    long long tot = (long long)n * c;
    f32_to_f16_k<<<dim3((unsigned)((tot + 255) / 256)), dim3(256), 0, stream>>>(
        src, n, c, cur, ld, coloff);
  };

  // ---- input: x -> f16 padded (n0+1) x 32 ----
  zero_h(stream, cur, (size_t)(n0 + 1) * 32);
  init_x16_k<<<dim3((unsigned)((n0 + 255) / 256)), dim3(256), 0, stream>>>(x, n0, cur);

  // ---- encoder ----
  cbr(32, 1, 1, 32, I(67), I(68), Ps1, n0, n0, nullptr, 0, S1, 32);     // conv1
  cbr(32, 4, 32, 32, I(67), I(68), Ps1, n0, n0, nullptr, 1, nullptr, 32); // b1a
  cbr(32, 7, 32, 32, I(67), I(68), Ps1, n0, n0, S1, 1, S1, 32);         // b1b (+res)
  cbr(32, 10, 32, 64, I(75), I(76), Pd1, n0, n1, nullptr, 0, S2, 64);   // conv2 (s2)
  cbr(64, 13, 64, 64, I(69), I(70), Ps2, n1, n1, nullptr, 1, nullptr, 64); // b2a
  cbr(64, 16, 64, 64, I(69), I(70), Ps2, n1, n1, S2, 1, S2, 64);        // b2b
  cbr(64, 19, 64, 128, I(77), I(78), Pd2, n1, n2, nullptr, 0, S4, 128); // conv3 (s4)
  cbr(128, 22, 128, 128, I(71), I(72), Ps4, n2, n2, nullptr, 1, nullptr, 128); // b3a
  cbr(128, 25, 128, 128, I(71), I(72), Ps4, n2, n2, S4, 1, S4, 128);    // b3b
  cbr(128, 28, 128, 256, I(79), I(80), Pd3, n2, n3, nullptr, 0, S8, 256); // conv4 (s8)
  cbr(256, 31, 256, 256, I(73), I(74), Ps8, n3, n3, nullptr, 1, nullptr, 256); // b4a
  cbr(256, 34, 256, 256, I(73), I(74), Ps8, n3, n3, S8, 1, S8, 256);    // b4b

  // ---- decoder (transpose convs = swapped kernel-map index arrays) ----
  cbr(256, 37, 256, 128, I(80), I(79), Pd3, n3, n2, nullptr, 0, T4, 128); // conv4_tr
  cbr(128, 40, 128, 128, I(71), I(72), Ps4, n2, n2, nullptr, 1, nullptr, 128); // b4ta
  cbr(128, 43, 128, 128, I(71), I(72), Ps4, n2, n2, T4, 1, nullptr, 256);  // b4tb -> cat
  concat(S4, n2, 128, 256, 128);                                           // [t4|s4]
  cbr(256, 46, 256, 64, I(78), I(77), Pd2, n2, n1, nullptr, 0, T2, 64);   // conv3_tr
  cbr(64, 49, 64, 64, I(69), I(70), Ps2, n1, n1, nullptr, 1, nullptr, 64); // b3ta
  cbr(64, 52, 64, 64, I(69), I(70), Ps2, n1, n1, T2, 1, nullptr, 128);    // b3tb -> cat
  concat(S2, n1, 64, 128, 64);                                             // [t2|s2]
  cbr(128, 55, 128, 64, I(76), I(75), Pd1, n1, n0, nullptr, 0, T1, 64);   // conv2_tr
  cbr(64, 58, 64, 64, I(67), I(68), Ps1, n0, n0, nullptr, 1, nullptr, 64); // b2ta
  cbr(64, 61, 64, 64, I(67), I(68), Ps1, n0, n0, T1, 1, nullptr, 96);     // b2tb -> cat
  concat(S1, n0, 32, 96, 64);                                              // [t1|s1]

  // ---- dense head: relu(t1cat @ W96x64) then @ W64x32 + b, then L2 norm ----
  {
    long long tw = 64LL * 96;
    cvt_w_dense_k<<<dim3((unsigned)((tw + 255) / 256)), dim3(256), 0, stream>>>(
        F(64), 96, 64, 96, W16);
    zero_h(stream, nxt, (size_t)(n0 + 1) * 64);
    dim3 g1((unsigned)((n0 + 63) / 64), 2);  // cout 64 -> two 32-wide tiles
    dgemm16_k<<<g1, dim3(128), (size_t)32 * 96 * sizeof(_Float16), stream>>>(
        cur, 96, W16, 64, n0, nullptr, 1, nullptr, nxt, 64);
    swapact();

    long long tw2 = 32LL * 64;
    cvt_w_dense_k<<<dim3((unsigned)((tw2 + 255) / 256)), dim3(256), 0, stream>>>(
        F(65), 64, 32, 64, W16);
    dim3 g2((unsigned)((n0 + 63) / 64), 1);  // cout 32 -> one 32-wide tile
    dgemm16_k<<<g2, dim3(128), (size_t)32 * 64 * sizeof(_Float16), stream>>>(
        cur, 64, W16, 32, n0, F(66), 0, TMP, (_Float16*)nullptr, 0);

    l2norm_k<<<dim3((unsigned)((n0 + 7) / 8)), dim3(256), 0, stream>>>(
        TMP, n0, (float*)d_out);
  }
}
